// Net_24429773979977
// MI455X (gfx1250) — matrix operand order repaired and checked by reading, not compiled, not run on hardware
//
#include <hip/hip_runtime.h>
#include <hip/hip_bf16.h>

typedef __attribute__((ext_vector_type(16))) _Float16 v16h;
typedef __attribute__((ext_vector_type(8)))  _Float16 v8h;
typedef __attribute__((ext_vector_type(4)))  _Float16 v4h;
typedef __attribute__((ext_vector_type(8)))  float    v8f;

#define CS_EPS 1e-8f

// Lane mapping chosen by this code (wave32; not the ISA table k order), V_WMMA_F32_16X16X32_F16:
//   A  v16h: lane (h,ln) holds row ln, halves k0+16h .. k0+16h+15 (contiguous)
//   B  v16h: lane (h,ln) holds col (ocol+ln), halves k0+16h .. k0+16h+15 (contiguous)
//   C/D v8f: VGPR r -> (M = r + 8h, N = ln)

// Load the A operand chunks for a 16-row tile once; reused across all output
// tiles of a layer (A is invariant in the N loop).
template <int KSTEPS>
__device__ __forceinline__ void load_a(const _Float16* A, int strideA,
                                       int ln, int h, v16h* a) {
#pragma unroll
  for (int ks = 0; ks < KSTEPS; ++ks) {
    const int k0 = ks * 32;
    v8h a0 = *(const v8h*)(A + ln * strideA + k0 + h * 16);
    v8h a1 = *(const v8h*)(A + ln * strideA + k0 + h * 16 + 8);
#pragma unroll
    for (int i = 0; i < 8; ++i) { a[ks][i] = a0[i]; a[ks][i + 8] = a1[i]; }
  }
}

// One 16x16 output tile: stream B chunks from LDS, accumulate KSTEPS WMMAs.
template <int KSTEPS>
__device__ __forceinline__ v8f mma_b(const v16h* a, const _Float16* Wt,
                                     int strideW, int ocol, int ln, int h) {
  v8f acc = {};
#pragma unroll
  for (int ks = 0; ks < KSTEPS; ++ks) {
    const int k0 = ks * 32;
    const _Float16* bp = Wt + (ocol + ln) * strideW + k0 + h * 16;
    v8h b0 = *(const v8h*)(bp);
    v8h b1 = *(const v8h*)(bp + 8);
    v16h b;
#pragma unroll
    for (int i = 0; i < 8; ++i) { b[i] = b0[i]; b[i + 8] = b1[i]; }
    acc = __builtin_amdgcn_wmma_f32_16x16x32_f16(
        false, a[ks], false, b, (short)0, acc, false, false);
  }
  return acc;
}

__global__ void __launch_bounds__(256)
mlp_cossim_kernel(const float* __restrict__ x,
                  const float* __restrict__ W1, const float* __restrict__ b1,
                  const float* __restrict__ W2, const float* __restrict__ b2,
                  const float* __restrict__ W3, const float* __restrict__ b3,
                  float* __restrict__ gAcc, int nRows) {
  extern __shared__ __align__(16) char smem[];
  _Float16* sW1 = (_Float16*)smem;          // [96][128]
  _Float16* sW2 = sW1 + 96 * 128;           // [80][96]  (N padded 72->80, zeros)
  _Float16* sW3 = sW2 + 80 * 96;            // [64][96]  (K padded 72->96, zeros)
  _Float16* sTA = sW3 + 64 * 96;            // 8 waves x [16][128]
  _Float16* sTB = sTA + 8 * 16 * 128;       // 8 waves x [16][128]
  float* sInv = (float*)(sTB + 8 * 16 * 128);  // 8 x 16 row inv-norms
  float* sS   = sInv + 8 * 16;                 // 64 column partials
  float* sUU  = sS + 64;                       // sum ||u||^2 partial

  const int tid = threadIdx.x;
  const int wave = tid >> 5, lane = tid & 31;
  const int h = lane >> 4, ln = lane & 15;

  // Stage weights fp32 -> f16 (with padding) into LDS, once per (persistent) block.
  for (int i = tid; i < 96 * 128; i += 256) sW1[i] = (_Float16)W1[i];
  for (int i = tid; i < 80 * 96; i += 256) {
    int o = i / 96, k = i - o * 96;
    sW2[i] = (_Float16)((o < 72) ? W2[o * 96 + k] : 0.0f);
  }
  for (int i = tid; i < 64 * 96; i += 256) {
    int o = i / 96, k = i - o * 96;
    sW3[i] = (_Float16)((k < 72) ? W3[o * 72 + k] : 0.0f);
  }
  if (tid < 64) sS[tid] = 0.0f;
  if (tid == 64) *sUU = 0.0f;
  __syncthreads();

  _Float16* TA = sTA + wave * (16 * 128);
  _Float16* TB = sTB + wave * (16 * 128);
  float* fT = (float*)TB;  // f32 [16][64] overlays TB once h1 is consumed
  float* inv = sInv + wave * 16;

  const int nTiles = nRows >> 4;
  const int tileStride = (int)gridDim.x * 8;

  // Persistent grid: each wave strides over 16-row tiles. Loop condition is
  // wave-uniform, so EXEC is all-1s inside (WMMA requirement). No barriers
  // needed inside the loop: per-wave LDS regions are private, sS/sUU use
  // LDS atomics.
  for (int t = (int)blockIdx.x * 8 + wave; t < nTiles; t += tileStride) {
    const int row0 = t << 4;

    // Stage x tile -> f16 LDS (each lane: 4 floats per row).
#pragma unroll
    for (int r = 0; r < 16; ++r) {
      const float4 v = ((const float4*)(x + (size_t)(row0 + r) * 128))[lane];
      v4h hv;
      hv[0] = (_Float16)v.x; hv[1] = (_Float16)v.y;
      hv[2] = (_Float16)v.z; hv[3] = (_Float16)v.w;
      *(v4h*)(TA + r * 128 + lane * 4) = hv;
    }

    // Prefetch next iteration's x tile (global_prefetch_b8): 16 rows x 512 B,
    // one prefetch per 256 B per lane.
    {
      const int tn = t + tileStride;
      if (tn < nTiles) {
        const float* p = x + (size_t)((tn << 4) + ln) * 128 + h * 64;
        __builtin_prefetch(p, 0, 3);
      }
    }

    // Layer 1: [16x128] @ [128x96] + b1, ReLU -> TB (f16, stride 128)
    {
      v16h a[4];
      load_a<4>(TA, 128, ln, h, a);
#pragma unroll
      for (int ot = 0; ot < 6; ++ot) {
        v8f acc = mma_b<4>(a, sW1, 128, ot * 16, ln, h);
        const float bv = b1[ot * 16 + ln];
#pragma unroll
        for (int r = 0; r < 8; ++r) {
          const int m = r + h * 8;
          TB[m * 128 + ot * 16 + ln] = (_Float16)fmaxf(acc[r] + bv, 0.0f);
        }
      }
    }

    // Layer 2: [16x96] @ [96x80pad] + b2, ReLU -> TA (f16, stride 128)
    {
      v16h a[3];
      load_a<3>(TB, 128, ln, h, a);
#pragma unroll
      for (int ot = 0; ot < 5; ++ot) {
        v8f acc = mma_b<3>(a, sW2, 96, ot * 16, ln, h);
        const int oc = ot * 16 + ln;
        const float bv = (oc < 72) ? b2[oc] : 0.0f;
#pragma unroll
        for (int r = 0; r < 8; ++r) {
          const int m = r + h * 8;
          TA[m * 128 + ot * 16 + ln] = (_Float16)fmaxf(acc[r] + bv, 0.0f);
        }
      }
    }
    // Zero K-pad columns 80..95 of h2 (rows covered by 32 lanes x 8 halves).
    { v8h z = {}; *(v8h*)(TA + ln * 128 + 80 + h * 8) = z; }

    // Layer 3: [16x96] @ [96x64] + b3 -> fT (f32, stride 64)
    {
      v16h a[3];
      load_a<3>(TA, 128, ln, h, a);
#pragma unroll
      for (int ot = 0; ot < 4; ++ot) {
        v8f acc = mma_b<3>(a, sW3, 96, ot * 16, ln, h);
        const float bv = b3[ot * 16 + ln];
#pragma unroll
        for (int r = 0; r < 8; ++r) {
          const int m = r + h * 8;
          fT[m * 64 + ot * 16 + ln] = acc[r] + bv;
        }
      }
    }

    // Row norms: lanes 0..15 each own one row.
    if (lane < 16) {
      float sq = 0.0f;
#pragma unroll 8
      for (int c = 0; c < 64; ++c) { float v = fT[ln * 64 + c]; sq += v * v; }
      const float iv = 1.0f / fmaxf(sqrtf(sq), CS_EPS);
      inv[ln] = iv;
      atomicAdd(sUU, sq * iv * iv);
    }

    // Column sums of u over this wave's 16 rows (lane -> cols c, c+32).
    {
      float c0 = 0.0f, c1 = 0.0f;
#pragma unroll
      for (int r = 0; r < 16; ++r) {
        const float iv = inv[r];
        c0 += fT[r * 64 + lane] * iv;
        c1 += fT[r * 64 + lane + 32] * iv;
      }
      atomicAdd(&sS[lane], c0);
      atomicAdd(&sS[lane + 32], c1);
    }
  }
  __syncthreads();

  // Block leaders flush partials to global accumulators (once per block).
  if (tid < 64) atomicAdd(&gAcc[tid], sS[tid]);
  if (tid == 64) atomicAdd(&gAcc[64], *sUU);
}

__global__ void zero_acc_kernel(float* g) {
  if (threadIdx.x < 65) g[threadIdx.x] = 0.0f;
}

__global__ void finalize_kernel(const float* __restrict__ g,
                                float* __restrict__ out, float invN) {
  const int l = threadIdx.x;  // 32 threads, one wave
  float v = g[l] * g[l] + g[l + 32] * g[l + 32];
#pragma unroll
  for (int off = 16; off > 0; off >>= 1) v += __shfl_down(v, off, 32);
  if (l == 0) out[0] = 0.5f * (v - g[64]) * invN;
}

extern "C" void kernel_launch(void* const* d_in, const int* in_sizes, int n_in,
                              void* d_out, int out_size, void* d_ws, size_t ws_size,
                              hipStream_t stream) {
  const float* x  = (const float*)d_in[0];
  const float* W1 = (const float*)d_in[1];
  const float* b1 = (const float*)d_in[2];
  const float* W2 = (const float*)d_in[3];
  const float* b2 = (const float*)d_in[4];
  const float* W3 = (const float*)d_in[5];
  const float* b3 = (const float*)d_in[6];
  float* gAcc = (float*)d_ws;   // 65 floats: s[64] + sumUU
  float* out  = (float*)d_out;

  const int nRows  = in_sizes[0] / 128;   // B*T
  const int nTiles = nRows / 16;
  int blocks = (nTiles + 7) / 8;          // 8 tiles (one per wave) per pass
  if (blocks > 1024) blocks = 1024;       // persistent grid; waves grid-stride

  const size_t smem =
      (size_t)(96 * 128 + 80 * 96 + 64 * 96 + 2 * 8 * 16 * 128) * sizeof(_Float16) +
      (size_t)(8 * 16 + 64 + 1) * sizeof(float);

  zero_acc_kernel<<<1, 96, 0, stream>>>(gAcc);
  mlp_cossim_kernel<<<blocks, 256, smem, stream>>>(x, W1, b1, W2, b2, W3, b3,
                                                   gAcc, nRows);
  finalize_kernel<<<1, 32, 0, stream>>>(gAcc, out, 1.0f / (float)nRows);
}
